// GPNN_HICO_2705829397266
// MI455X (gfx1250) — compile-verified, matrix operand order checked by reading, not run on hardware
//
#include <hip/hip_runtime.h>
#include <hip/hip_bf16.h>

// CDNA5 (gfx1250) fused GPNN forward.
// Heavy GEMMs on v_wmma_f32_16x16x32_bf16. Weights staged once per block in
// LDS, pre-swizzled into B-fragment order so each fragment is one 32B LDS
// vector load. A fragments are 128-bit global vector loads. Gate scaling is
// applied post-WMMA on the f32 accumulators (commutes with the matmul).

typedef __attribute__((ext_vector_type(16))) __bf16 v16bf;
typedef __attribute__((ext_vector_type(8)))  __bf16 v8bf;
typedef __attribute__((ext_vector_type(8)))  float  v8f;

#define BB    4
#define NN    256
#define EFD   256
#define MSGF  128
#define NCLS  600
#define ETOT  (BB*NN*NN)   // 262144 edges

static __device__ inline float sigmf(float x) { return 1.0f / (1.0f + __expf(-x)); }

static __device__ inline v8f wmma_bf16(v16bf a, v16bf b, v8f c) {
  return __builtin_amdgcn_wmma_f32_16x16x32_bf16(
      /*neg_a=*/false, a, /*neg_b=*/false, b,
      /*c_mod=*/(short)0, c, /*reuse_a=*/false, /*reuse_b=*/false);
}

// A fragment (16x32 bf16): lane = 16*khalf + m holds k in [kb,kb+8)+[16+kb,16+kb+8).
static __device__ inline v16bf load_a_bf16(const __bf16* row, int kb) {
  v8bf lo = *(const v8bf*)(row + kb);        // 16B global_load_b128
  v8bf hi = *(const v8bf*)(row + 16 + kb);   // 16B global_load_b128
  return __builtin_shufflevector(lo, hi, 0,1,2,3,4,5,6,7,8,9,10,11,12,13,14,15);
}

static __device__ inline v16bf load_a_f32(const float* row, int kb) {
  const float4 a0 = *(const float4*)(row + kb);
  const float4 a1 = *(const float4*)(row + kb + 4);
  const float4 b0 = *(const float4*)(row + 16 + kb);
  const float4 b1 = *(const float4*)(row + 16 + kb + 4);
  v16bf f;
  f[0] = (__bf16)a0.x; f[1] = (__bf16)a0.y; f[2] = (__bf16)a0.z; f[3] = (__bf16)a0.w;
  f[4] = (__bf16)a1.x; f[5] = (__bf16)a1.y; f[6] = (__bf16)a1.z; f[7] = (__bf16)a1.w;
  f[8]  = (__bf16)b0.x; f[9]  = (__bf16)b0.y; f[10] = (__bf16)b0.z; f[11] = (__bf16)b0.w;
  f[12] = (__bf16)b1.x; f[13] = (__bf16)b1.y; f[14] = (__bf16)b1.z; f[15] = (__bf16)b1.w;
  return f;
}

// Stage one 32xMSGF fp32 weight chunk into LDS, swizzled into B-fragment order:
// layout [tile tl][lane L][16 bf16]. Thread t owns fragment slot (tl=t>>5, L=t&31).
static __device__ inline void stage_w_chunk(__bf16* sdst, const float* W, int row0) {
  const int t  = threadIdx.x;
  const int tl = t >> 5, L = t & 31;
  const int n  = tl * 16 + (L & 15);
  const int kb = (L >> 4) << 3;
  __bf16* dst = sdst + (((tl << 5) + L) << 4);
#pragma unroll
  for (int e = 0; e < 8; ++e) {
    dst[e]     = (__bf16)W[(row0 + kb + e) * MSGF + n];
    dst[e + 8] = (__bf16)W[(row0 + 16 + kb + e) * MSGF + n];
  }
}

// Fetch a pre-swizzled B fragment: one aligned 32B LDS read.
static __device__ inline v16bf load_b_lds(const __bf16* sW, int s, int tl, int lane) {
  return *(const v16bf*)(sW + (s << 12) + (((tl << 5) + lane) << 4));
}

// ---------------------------------------------------------------------------
// nf = node @ W_nr + b_nr (fp32 for the GRU), nfm = nf @ W_msg[:128] + b_msg
__global__ __launch_bounds__(128) void k_node(
    const float* __restrict__ node, const float* __restrict__ Wnr,
    const float* __restrict__ bnr, const float* __restrict__ Wmsg,
    const float* __restrict__ bmsg, float* __restrict__ nf,
    float* __restrict__ nfm) {
  __shared__ float snf[MSGF];
  const int row = blockIdx.x, j = threadIdx.x;
  const float* x = node + (size_t)row * EFD;
  float a = bnr[j];
  for (int k = 0; k < EFD; ++k) a = fmaf(x[k], Wnr[k * MSGF + j], a);
  nf[(size_t)row * MSGF + j] = a;
  snf[j] = a;
  __syncthreads();
  float a2 = bmsg[j];
  for (int k = 0; k < MSGF; ++k) a2 = fmaf(snf[k], Wmsg[k * MSGF + j], a2);
  nfm[(size_t)row * MSGF + j] = a2;
}

// ---------------------------------------------------------------------------
// ef_bf16 = bf16(edge_features @ W_er + b_er)   [ETOT x 128], K=256
__global__ __launch_bounds__(256) void k_edge_resize(
    const float* __restrict__ edge, const float* __restrict__ Wer,
    const float* __restrict__ ber, __bf16* __restrict__ ef) {
  __shared__ __bf16 sW[8 * 4096];                 // 64 KB, full 256x128 swizzled
  const int t = threadIdx.x, lane = t & 31, wave = t >> 5;
  const int ebase = blockIdx.x * 128 + wave * 16;
  const int ma = lane & 15, kb = (lane >> 4) << 3;
  const float* arow = edge + (size_t)(ebase + ma) * EFD;

#pragma unroll
  for (int s = 0; s < 8; ++s) stage_w_chunk(sW + (s << 12), Wer, s * 32);
  __syncthreads();

  v8f acc[8];
#pragma unroll
  for (int tl = 0; tl < 8; ++tl) acc[tl] = (v8f){0,0,0,0,0,0,0,0};

  for (int s = 0; s < 8; ++s) {
    v16bf a = load_a_f32(arow + s * 32, kb);
#pragma unroll
    for (int tl = 0; tl < 8; ++tl)
      acc[tl] = wmma_bf16(a, load_b_lds(sW, s, tl, lane), acc[tl]);
  }

  const int mc = (lane >> 4) << 3, nlo = lane & 15;
#pragma unroll
  for (int tl = 0; tl < 8; ++tl) {
    const int n = tl * 16 + nlo;
    const float bias = ber[n];
#pragma unroll
    for (int r = 0; r < 8; ++r)
      ef[(size_t)(ebase + r + mc) * MSGF + n] = (__bf16)(acc[tl][r] + bias);
  }
}

// ---------------------------------------------------------------------------
// m_raw = bf16(relu(nfm[b,w] + ef @ W_msg[128:256]))    K=128
__global__ __launch_bounds__(256) void k_mraw(
    const __bf16* __restrict__ ef, const float* __restrict__ Wmsg,
    const float* __restrict__ nfm, __bf16* __restrict__ mraw) {
  __shared__ __bf16 sW[4 * 4096];                 // 32 KB
  const int t = threadIdx.x, lane = t & 31, wave = t >> 5;
  const int ebase = blockIdx.x * 128 + wave * 16;
  const int ma = lane & 15, kb = (lane >> 4) << 3;
  const __bf16* arow = ef + (size_t)(ebase + ma) * MSGF;

#pragma unroll
  for (int s = 0; s < 4; ++s) stage_w_chunk(sW + (s << 12), Wmsg, 128 + s * 32);
  __syncthreads();

  v8f acc[8];
#pragma unroll
  for (int tl = 0; tl < 8; ++tl) acc[tl] = (v8f){0,0,0,0,0,0,0,0};

#pragma unroll
  for (int s = 0; s < 4; ++s) {
    v16bf a = load_a_bf16(arow + s * 32, kb);
#pragma unroll
    for (int tl = 0; tl < 8; ++tl)
      acc[tl] = wmma_bf16(a, load_b_lds(sW, s, tl, lane), acc[tl]);
  }

  const int mc = (lane >> 4) << 3, nlo = lane & 15;
#pragma unroll
  for (int tl = 0; tl < 8; ++tl) {
    const int n = tl * 16 + nlo;
#pragma unroll
    for (int r = 0; r < 8; ++r) {
      const int e = ebase + r + mc;
      const int b = e >> 16, w = e & 255;
      float v = acc[tl][r] + nfm[(size_t)(b * NN + w) * MSGF + n];
      mraw[(size_t)e * MSGF + n] = (__bf16)fmaxf(v, 0.0f);
    }
  }
}

// ---------------------------------------------------------------------------
// One link round: pred = relu(e_state @ W_l1 + b_l1) @ W_l2 + b_l2, masked.
// e_state rows gathered as: round 0 -> ef[i,w]; round>0 ->
//   pair ? m_raw[w,i] : ef[i,w]. The gate factor (per row, pair only) is
//   applied to the f32 accumulators post-WMMA (commutes with the matmul).
__global__ __launch_bounds__(256) void k_link(
    const __bf16* __restrict__ ef, const __bf16* __restrict__ mraw,
    const float* __restrict__ gate_prev, const float* __restrict__ Wl1,
    const float* __restrict__ bl1, const float* __restrict__ Wl2,
    const float* __restrict__ bl2, const int* __restrict__ hnum,
    const int* __restrict__ onum, float* __restrict__ gate_out,
    float* __restrict__ adj_out, int round) {
  __shared__ __bf16 sW[4 * 4096];                 // 32 KB
  const int t = threadIdx.x, lane = t & 31, wave = t >> 5;
  const int ebase = blockIdx.x * 128 + wave * 16;
  const int b = ebase >> 16, i = (ebase >> 8) & 255, wbase = ebase & 255;
  const int valid = hnum[b] + onum[b];
  const int ma = lane & 15, kb = (lane >> 4) << 3;

  const int wa = wbase + ma;
  const bool pair_a = (round > 0) && (i < valid) && (wa < valid);
  const __bf16* arow = pair_a
      ? mraw + ((size_t)(b * NN + wa) * NN + i) * MSGF    // transposed edge (w,i)
      : ef + ((size_t)ebase + ma) * MSGF;

#pragma unroll
  for (int s = 0; s < 4; ++s) stage_w_chunk(sW + (s << 12), Wl1, s * 32);
  __syncthreads();

  v8f acc[8];
#pragma unroll
  for (int tl = 0; tl < 8; ++tl) acc[tl] = (v8f){0,0,0,0,0,0,0,0};

#pragma unroll
  for (int s = 0; s < 4; ++s) {
    v16bf a = load_a_bf16(arow + s * 32, kb);
#pragma unroll
    for (int tl = 0; tl < 8; ++tl)
      acc[tl] = wmma_bf16(a, load_b_lds(sW, s, tl, lane), acc[tl]);
  }

  const int nlo = lane & 15, mc = (lane >> 4) << 3;
  float bl1n[8], wl2n[8];
#pragma unroll
  for (int tl = 0; tl < 8; ++tl) {
    bl1n[tl] = bl1[tl * 16 + nlo];
    wl2n[tl] = Wl2[tl * 16 + nlo];
  }
  // Per-C-row gate factor (C layout: row m = r + mc).
  float rsc[8];
#pragma unroll
  for (int r = 0; r < 8; ++r) {
    const int w = wbase + r + mc;
    rsc[r] = ((round > 0) && (i < valid) && (w < valid))
                 ? gate_prev[(size_t)(b * NN + w) * NN + i]
                 : 1.0f;
  }
  const float bias2 = bl2[0];
#pragma unroll
  for (int r = 0; r < 8; ++r) {
    float p = 0.0f;
#pragma unroll
    for (int tl = 0; tl < 8; ++tl)
      p += fmaxf(fmaf(rsc[r], acc[tl][r], bl1n[tl]), 0.0f) * wl2n[tl];
    p += __shfl_xor(p, 1, 32);
    p += __shfl_xor(p, 2, 32);
    p += __shfl_xor(p, 4, 32);
    p += __shfl_xor(p, 8, 32);
    if (nlo == 0) {                       // lanes 0 and 16 own m=r and m=r+8
      const int m = r + mc, w = wbase + m;
      const size_t e = (size_t)ebase + m;
      const bool pair = (i < valid) && (w < valid);
      const float adj = pair ? (p + bias2) : 0.0f;
      gate_out[e] = sigmf(adj) * ((w < valid) ? 1.0f : 0.0f);
      if (adj_out) adj_out[e] = adj;
    }
  }
}

// ---------------------------------------------------------------------------
// m_sum[b,i,c] = sum_w gate2[b,i,w] * m_raw[b,i,w,c]
__global__ __launch_bounds__(128) void k_msum(
    const __bf16* __restrict__ mraw, const float* __restrict__ gate,
    float* __restrict__ msum) {
  __shared__ float sg[NN];
  const int row = blockIdx.x, c = threadIdx.x;
  for (int w = c; w < NN; w += 128) sg[w] = gate[(size_t)row * NN + w];
  __syncthreads();
  const __bf16* base = mraw + (size_t)row * NN * MSGF + c;
  float a = 0.0f;
  for (int w = 0; w < NN; ++w) a = fmaf(sg[w], (float)base[(size_t)w * MSGF], a);
  msum[(size_t)row * MSGF + c] = a;
}

// ---------------------------------------------------------------------------
// GRU (torch gate order r,z,n) + readout, node-masked.
__global__ __launch_bounds__(128) void k_gru_readout(
    const float* __restrict__ msum, const float* __restrict__ nf,
    const float* __restrict__ Wih, const float* __restrict__ bih,
    const float* __restrict__ Whh, const float* __restrict__ bhh,
    const float* __restrict__ Wro, const float* __restrict__ bro,
    const int* __restrict__ hnum, const int* __restrict__ onum,
    float* __restrict__ labels) {
  __shared__ float sx[MSGF], sh[MSGF], shn[MSGF];
  const int row = blockIdx.x, j = threadIdx.x;
  const int b = row >> 8, i = row & 255;
  const int valid = hnum[b] + onum[b];
  sx[j] = msum[(size_t)row * MSGF + j];
  sh[j] = nf[(size_t)row * MSGF + j];
  __syncthreads();
  float gi[3], gh[3];
#pragma unroll
  for (int g = 0; g < 3; ++g) {
    const int col = g * MSGF + j;
    float a = bih[col], c = bhh[col];
    for (int k = 0; k < MSGF; ++k) {
      a = fmaf(sx[k], Wih[k * 384 + col], a);
      c = fmaf(sh[k], Whh[k * 384 + col], c);
    }
    gi[g] = a; gh[g] = c;
  }
  const float rg = sigmf(gi[0] + gh[0]);
  const float zg = sigmf(gi[1] + gh[1]);
  const float ng = tanhf(gi[2] + rg * gh[2]);
  shn[j] = (1.0f - zg) * ng + zg * sh[j];
  __syncthreads();
  const float mask = (i < valid) ? 1.0f : 0.0f;
  for (int o = j; o < NCLS; o += 128) {
    float a = bro[o];
    for (int k = 0; k < MSGF; ++k) a = fmaf(shn[k], Wro[k * NCLS + o], a);
    labels[(size_t)row * NCLS + o] = mask * a;
  }
}

// ---------------------------------------------------------------------------
extern "C" void kernel_launch(void* const* d_in, const int* in_sizes, int n_in,
                              void* d_out, int out_size, void* d_ws, size_t ws_size,
                              hipStream_t stream) {
  (void)in_sizes; (void)n_in; (void)out_size; (void)ws_size;
  const float* edge = (const float*)d_in[0];
  const float* node = (const float*)d_in[1];
  const int*   hnum = (const int*)d_in[4];
  const int*   onum = (const int*)d_in[5];
  const float* Wer  = (const float*)d_in[6];
  const float* ber  = (const float*)d_in[7];
  const float* Wnr  = (const float*)d_in[8];
  const float* bnr  = (const float*)d_in[9];
  const float* Wl1  = (const float*)d_in[10];
  const float* bl1  = (const float*)d_in[11];
  const float* Wl2  = (const float*)d_in[12];
  const float* bl2  = (const float*)d_in[13];
  const float* Wmsg = (const float*)d_in[14];
  const float* bmsg = (const float*)d_in[15];
  const float* Wih  = (const float*)d_in[16];
  const float* bih  = (const float*)d_in[17];
  const float* Whh  = (const float*)d_in[18];
  const float* bhh  = (const float*)d_in[19];
  const float* Wro  = (const float*)d_in[20];
  const float* bro  = (const float*)d_in[21];

  char* ws = (char*)d_ws;
  __bf16* ef    = (__bf16*)(ws);                       // 67,108,864 B
  __bf16* mraw  = (__bf16*)(ws + 67108864);            // 67,108,864 B
  float*  nf    = (float*)(ws + 134217728);            //    524,288 B
  float*  nfm   = (float*)(ws + 134742016);            //    524,288 B
  float*  msum  = (float*)(ws + 135266304);            //    524,288 B
  float*  gate0 = (float*)(ws + 135790592);            //  1,048,576 B
  float*  gate1 = (float*)(ws + 136839168);            //  1,048,576 B

  float* adj_out = (float*)d_out;                      // [B,N,N]
  float* lbl_out = (float*)d_out + ETOT;               // [B,N,CLASSES]

  const int eblocks = ETOT / 128;                      // 2048

  k_node<<<dim3(BB * NN), dim3(128), 0, stream>>>(node, Wnr, bnr, Wmsg, bmsg, nf, nfm);
  k_edge_resize<<<dim3(eblocks), dim3(256), 0, stream>>>(edge, Wer, ber, ef);
  k_mraw<<<dim3(eblocks), dim3(256), 0, stream>>>(ef, Wmsg, nfm, mraw);

  // round 0: e_state = ef
  k_link<<<dim3(eblocks), dim3(256), 0, stream>>>(ef, mraw, gate0, Wl1, bl1, Wl2, bl2,
                                                  hnum, onum, gate0, (float*)nullptr, 0);
  // round 1
  k_link<<<dim3(eblocks), dim3(256), 0, stream>>>(ef, mraw, gate0, Wl1, bl1, Wl2, bl2,
                                                  hnum, onum, gate1, (float*)nullptr, 1);
  // round 2 (writes pred_adj)
  k_link<<<dim3(eblocks), dim3(256), 0, stream>>>(ef, mraw, gate1, Wl1, bl1, Wl2, bl2,
                                                  hnum, onum, gate0, adj_out, 2);

  k_msum<<<dim3(BB * NN), dim3(128), 0, stream>>>(mraw, gate0, msum);
  k_gru_readout<<<dim3(BB * NN), dim3(128), 0, stream>>>(msum, nf, Wih, bih, Whh, bhh,
                                                         Wro, bro, hnum, onum, lbl_out);
}
